// block_74955769250655
// MI455X (gfx1250) — matrix operand order repaired and checked by reading, not compiled, not run on hardware
//
#include <hip/hip_runtime.h>
#include <hip/hip_bf16.h>

typedef __attribute__((ext_vector_type(16))) _Float16 v16h;
typedef __attribute__((ext_vector_type(8)))  float    v8f;
typedef __attribute__((ext_vector_type(4)))  int      i32x4;

#define BB   32
#define SS   512
#define DD   256
#define HH   8
#define EE   32
#define FFD  1024
#define LN_EPS 1e-5f

// CDNA5 async global->LDS copy path (guarded: falls back to direct loads)
#if defined(__has_builtin)
#if __has_builtin(__builtin_amdgcn_global_load_async_to_lds_b128) && \
    __has_builtin(__builtin_amdgcn_s_wait_asynccnt)
#define HAVE_ASYNC_LDS 1
#endif
#endif
#ifndef HAVE_ASYNC_LDS
#define HAVE_ASYNC_LDS 0
#endif

#if HAVE_ASYNC_LDS
typedef __attribute__((address_space(1))) i32x4 as1_i32x4;  // global (prints __device__)
typedef __attribute__((address_space(3))) i32x4 as3_i32x4;  // LDS    (prints __shared__)
#endif

// ---------------- fragment-swizzled tile layouts -----------------------------
// A tile: 16(m) x 32(k) f16, 512 halves. Lane l holds 16 contiguous halves that
// are exactly its A-fragment registers (documented gfx1250 wave32 layout).
__device__ __forceinline__ int a_swz_idx(int m, int k) {
  return (m + 16 * ((k >> 3) & 1)) * 16 + (k & 7) + 8 * ((k >> 4) & 1);
}
// B tile: 32(k) x 16(n) f16, 512 halves, same per-lane-contiguous property.
__device__ __forceinline__ int b_swz_idx(int k, int n) {
  return (n + 16 * ((k >> 3) & 1)) * 16 + (k & 7) + 8 * ((k >> 4) & 1);
}

// One fragment = 2 x 16-byte vector loads per lane (coalesced 1KB per wave).
__device__ __forceinline__ v16h load_frag_swz(const _Float16* tile) {
  const int lane = threadIdx.x & 31;
  union { v16h h; uint4 q[2]; } r;
  const uint4* p = (const uint4*)(tile + lane * 16);
  r.q[0] = p[0];
  r.q[1] = p[1];
  return r.h;
}

__device__ __forceinline__ v8f zero8() {
  v8f z;
#pragma unroll
  for (int i = 0; i < 8; ++i) z[i] = 0.0f;
  return z;
}

__device__ __forceinline__ v8f wmma16(v16h a, v16h b, v8f c) {
  return __builtin_amdgcn_wmma_f32_16x16x32_f16(false, a, false, b,
                                                (short)0, c, false, false);
}

// 16-lane (half-wave) reductions; xor masks 1..8 never cross the half boundary
__device__ __forceinline__ float rowmax16(float x) {
#pragma unroll
  for (int m = 8; m >= 1; m >>= 1) x = fmaxf(x, __shfl_xor(x, m, 32));
  return x;
}
__device__ __forceinline__ float rowsum16(float x) {
#pragma unroll
  for (int m = 8; m >= 1; m >>= 1) x += __shfl_xor(x, m, 32);
  return x;
}

// ---------------- conversion kernels (f32 row-major -> f16 swizzled) ---------

__global__ void cvt_a_swz_kernel(const float* __restrict__ src,
                                 _Float16* __restrict__ dst, int C, int n) {
  int i = blockIdx.x * blockDim.x + threadIdx.x;
  if (i >= n) return;
  int tile = i >> 9, w = i & 511;
  int l = w >> 4, c = w & 15;
  int ntk = C >> 5;
  int tr = tile / ntk, tk = tile - tr * ntk;
  int m = l & 15;
  int k = (c & 7) + 8 * (l >> 4) + 16 * (c >> 3);
  dst[i] = (_Float16)src[(tr * 16 + m) * C + tk * 32 + k];
}

__global__ void cvt_b_swz_kernel(const float* __restrict__ src,
                                 _Float16* __restrict__ dst, int N, int n) {
  int i = blockIdx.x * blockDim.x + threadIdx.x;
  if (i >= n) return;
  int tile = i >> 9, w = i & 511;
  int l = w >> 4, c = w & 15;
  int ntn = N >> 4;
  int tk = tile / ntn, tn = tile - tk * ntn;
  int nn = l & 15;
  int k = (c & 7) + 8 * (l >> 4) + 16 * (c >> 3);
  dst[i] = (_Float16)src[(tk * 32 + k) * N + tn * 16 + nn];
}

// ---------------- QKV projection --------------------------------------------
__global__ void qkv_kernel(const _Float16* __restrict__ xh,
                           const _Float16* __restrict__ wqh,
                           const _Float16* __restrict__ wkh,
                           const _Float16* __restrict__ wvh,
                           _Float16* __restrict__ qh,
                           _Float16* __restrict__ kh,
                           _Float16* __restrict__ vh) {
  const int gw = blockIdx.x * (blockDim.x >> 5) + (threadIdx.x >> 5);
  const int b  = gw >> 8;
  const int h  = (gw >> 5) & 7;
  const int rb = gw & 31;
  const int lane = threadIdx.x & 31;
  const int n = lane & 15, hf = lane >> 4;
  const int tr = b * 32 + rb;
  const _Float16* wq = wqh + h * DD * EE;
  const _Float16* wk = wkh + h * DD * EE;
  const _Float16* wv = wvh + h * DD * EE;
  v8f q0 = zero8(), q1 = zero8(), k0 = zero8(), k1 = zero8(), v0 = zero8(), v1 = zero8();
#pragma unroll
  for (int kk = 0; kk < 8; ++kk) {
    v16h a  = load_frag_swz(xh + (size_t)(tr * 8 + kk) * 512);
    v16h b0 = load_frag_swz(wq + (kk * 2 + 0) * 512);
    v16h b1 = load_frag_swz(wq + (kk * 2 + 1) * 512);
    q0 = wmma16(a, b0, q0); q1 = wmma16(a, b1, q1);
    b0 = load_frag_swz(wk + (kk * 2 + 0) * 512);
    b1 = load_frag_swz(wk + (kk * 2 + 1) * 512);
    k0 = wmma16(a, b0, k0); k1 = wmma16(a, b1, k1);
    b0 = load_frag_swz(wv + (kk * 2 + 0) * 512);
    b1 = load_frag_swz(wv + (kk * 2 + 1) * 512);
    v0 = wmma16(a, b0, v0); v1 = wmma16(a, b1, v1);
  }
  const size_t hd = (size_t)(b * HH + h) * SS * EE;
  _Float16* qd = qh + hd + rb * 512;                // A-swz tile rb
  _Float16* kd = kh + hd + rb * 512;                // B-swz tile (n-block rb)
  _Float16* vd = vh + hd;                           // B-swz tiles (k = ctx pos)
#pragma unroll
  for (int v = 0; v < 8; ++v) {
    const int r = v + 8 * hf;
    qd[a_swz_idx(r, n)]      = (_Float16)q0[v];
    qd[a_swz_idx(r, 16 + n)] = (_Float16)q1[v];
    kd[b_swz_idx(n, r)]      = (_Float16)k0[v];
    kd[b_swz_idx(16 + n, r)] = (_Float16)k1[v];
    const int kin = ((rb & 1) << 4) + r;
    vd[(size_t)((rb >> 1) * 2 + 0) * 512 + b_swz_idx(kin, n)] = (_Float16)v0[v];
    vd[(size_t)((rb >> 1) * 2 + 1) * 512 + b_swz_idx(kin, n)] = (_Float16)v1[v];
  }
}

// ---------------- flash-style causal attention -------------------------------
__global__ void attn_kernel(const _Float16* __restrict__ qh,
                            const _Float16* __restrict__ kh,
                            const _Float16* __restrict__ vh,
                            _Float16* __restrict__ ctxh) {
  __shared__ __align__(16) _Float16 lds_p[4 * 512];  // 1 KB per wave
  const int wid = threadIdx.x >> 5;
  const int gw  = blockIdx.x * 4 + wid;
  const int b   = gw >> 8;
  const int h   = (gw >> 5) & 7;
  const int rb  = gw & 31;
  const int lane = threadIdx.x & 31;
  const int n = lane & 15, hf = lane >> 4;
  const size_t hd = (size_t)(b * HH + h) * SS * EE;
  const _Float16* kb = kh + hd;
  const _Float16* vb = vh + hd;
  v16h qfrag = load_frag_swz(qh + hd + rb * 512);
  float mrow[8], lrow[8];
#pragma unroll
  for (int v = 0; v < 8; ++v) { mrow[v] = -1e30f; lrow[v] = 0.0f; }
  v8f acc0 = zero8(), acc1 = zero8();
  const float scale = 0.17677669529663687f;         // 1/sqrt(32)
  const int nj = (rb * 16 + 15) / 32 + 1;           // causal pruning
  _Float16* pl = lds_p + wid * 512;
  for (int j = 0; j < nj; ++j) {
    const int c0 = j * 32;
    v16h bt0 = load_frag_swz(kb + (2 * j + 0) * 512);
    v16h bt1 = load_frag_swz(kb + (2 * j + 1) * 512);
    v8f s0 = wmma16(qfrag, bt0, zero8());           // K = E = 32: single WMMA
    v8f s1 = wmma16(qfrag, bt1, zero8());
    float p0[8], p1[8];
#pragma unroll
    for (int v = 0; v < 8; ++v) {
      const int row = rb * 16 + v + 8 * hf;
      float a0 = (c0 + n      <= row) ? s0[v] * scale : -1e30f;
      float a1 = (c0 + 16 + n <= row) ? s1[v] * scale : -1e30f;
      float mn = fmaxf(mrow[v], rowmax16(fmaxf(a0, a1)));
      float alpha = __expf(mrow[v] - mn);
      mrow[v] = mn;
      float e0 = __expf(a0 - mn), e1 = __expf(a1 - mn);
      lrow[v] = lrow[v] * alpha + rowsum16(e0 + e1);
      acc0[v] *= alpha; acc1[v] *= alpha;
      p0[v] = e0; p1[v] = e1;
    }
    // repack P (C layout) -> A-swizzled LDS tile (in-order DS within a wave)
#pragma unroll
    for (int v = 0; v < 8; ++v) {
      const int r = v + 8 * hf;
      pl[a_swz_idx(r, n)]      = (_Float16)p0[v];
      pl[a_swz_idx(r, 16 + n)] = (_Float16)p1[v];
    }
    v16h pf  = load_frag_swz(pl);                   // 2 x ds_load_b128
    v16h bv0 = load_frag_swz(vb + (j * 2 + 0) * 512);
    v16h bv1 = load_frag_swz(vb + (j * 2 + 1) * 512);
    acc0 = wmma16(pf, bv0, acc0);
    acc1 = wmma16(pf, bv1, acc1);
  }
  _Float16* cd = ctxh + (size_t)((b * 32 + rb) * 8 + h) * 512;
#pragma unroll
  for (int v = 0; v < 8; ++v) {
    const float inv = 1.0f / lrow[v];
    const int r = v + 8 * hf;
    cd[a_swz_idx(r, n)]      = (_Float16)(acc0[v] * inv);
    cd[a_swz_idx(r, 16 + n)] = (_Float16)(acc1[v] * inv);
  }
}

// ------- cooperative double-buffered weight staging (async global->LDS) -----
// Stages 16 B-tiles (16KB) per K-step; all 4 waves of the block consume them.
#if HAVE_ASYNC_LDS
__device__ __forceinline__ void stage_weights(const _Float16* __restrict__ src,
                                              _Float16* __restrict__ dst) {
  const int tid = threadIdx.x;
#pragma unroll
  for (int t = 0; t < 8; ++t) {
    const int off = (t * 128 + tid) * 8;  // 16B segments, 16KB total per block
    __builtin_amdgcn_global_load_async_to_lds_b128(
        (as1_i32x4*)(src + off), (as3_i32x4*)(dst + off), 0, 0);
  }
}
#endif

// ---------------- proj + residual + LN1 --------------------------------------
__global__ void proj_ln_kernel(const _Float16* __restrict__ ctxh,
                               const _Float16* __restrict__ wph,
                               const float* __restrict__ x,
                               const float* __restrict__ g,
                               const float* __restrict__ be,
                               _Float16* __restrict__ hh) {
  const int tr = blockIdx.x * 4 + (threadIdx.x >> 5);   // 16-row block, 0..1023
  const int lane = threadIdx.x & 31;
  const int n = lane & 15, hf = lane >> 4;
  v8f acc[16];
#pragma unroll
  for (int f = 0; f < 16; ++f) acc[f] = zero8();
#if HAVE_ASYNC_LDS
  __shared__ __align__(16) _Float16 wbuf[2][16 * 512];  // 2 x 16KB
  stage_weights(wph, wbuf[0]);
  for (int kk = 0; kk < 8; ++kk) {
    const int buf = kk & 1;
    v16h a = load_frag_swz(ctxh + (size_t)(tr * 8 + kk) * 512);
    __builtin_amdgcn_s_wait_asynccnt(0);
    __syncthreads();
    if (kk + 1 < 8) stage_weights(wph + (size_t)(kk + 1) * 16 * 512, wbuf[buf ^ 1]);
#pragma unroll
    for (int f = 0; f < 16; ++f)
      acc[f] = wmma16(a, load_frag_swz(&wbuf[buf][f * 512]), acc[f]);
    __syncthreads();
  }
#else
  for (int kk = 0; kk < 8; ++kk) {
    if (kk + 1 < 8) __builtin_prefetch(wph + (size_t)(kk + 1) * 16 * 512, 0, 0);
    v16h a = load_frag_swz(ctxh + (size_t)(tr * 8 + kk) * 512);
#pragma unroll
    for (int f = 0; f < 16; ++f)
      acc[f] = wmma16(a, load_frag_swz(wph + (size_t)(kk * 16 + f) * 512), acc[f]);
  }
#endif
#pragma unroll
  for (int v = 0; v < 8; ++v) {
    float s = 0.f, q = 0.f;
#pragma unroll
    for (int f = 0; f < 16; ++f) { float t = acc[f][v]; s += t; q += t * t; }
    s = rowsum16(s); q = rowsum16(q);
    const float mean = s * (1.0f / DD);
    const float var  = q * (1.0f / DD) - mean * mean;
    const float rstd = rsqrtf(var + LN_EPS);
    const int r = v + 8 * hf;
    const int row = tr * 16 + r;
#pragma unroll
    for (int f = 0; f < 16; ++f) {
      const int col = f * 16 + n;
      float val = (acc[f][v] - mean) * rstd * g[col] + be[col] + x[row * DD + col];
      hh[(size_t)(tr * 8 + (f >> 1)) * 512 + a_swz_idx(r, ((f & 1) << 4) + n)] =
          (_Float16)val;
    }
  }
}

// ---------------- FF1: relu(h @ w1 + b1) -------------------------------------
__global__ void ff1_kernel(const _Float16* __restrict__ hh,
                           const _Float16* __restrict__ w1h,
                           const float* __restrict__ b1,
                           _Float16* __restrict__ a1h) {
  const int gw = blockIdx.x * 4 + (threadIdx.x >> 5);
  const int tr = gw >> 4;
  const int cb = gw & 15;                              // 64-col block
  const int lane = threadIdx.x & 31;
  const int n = lane & 15, hf = lane >> 4;
  v8f acc[4];
#pragma unroll
  for (int f = 0; f < 4; ++f) acc[f] = zero8();
  for (int kk = 0; kk < 8; ++kk) {
    v16h a = load_frag_swz(hh + (size_t)(tr * 8 + kk) * 512);
#pragma unroll
    for (int f = 0; f < 4; ++f) {
      v16h bw = load_frag_swz(w1h + (size_t)(kk * 64 + cb * 4 + f) * 512);
      acc[f] = wmma16(a, bw, acc[f]);
    }
  }
#pragma unroll
  for (int v = 0; v < 8; ++v) {
    const int r = v + 8 * hf;
#pragma unroll
    for (int f = 0; f < 4; ++f) {
      const int col = cb * 64 + f * 16 + n;
      float t = fmaxf(acc[f][v] + b1[col], 0.0f);
      a1h[(size_t)(tr * 32 + cb * 2 + (f >> 1)) * 512 +
          a_swz_idx(r, ((f & 1) << 4) + n)] = (_Float16)t;
    }
  }
}

// ---------------- FF2 + bias + LN2 + residual --------------------------------
__global__ void ff2_ln_kernel(const _Float16* __restrict__ a1h,
                              const _Float16* __restrict__ w2h,
                              const float* __restrict__ b2,
                              const float* __restrict__ x,
                              const float* __restrict__ g,
                              const float* __restrict__ be,
                              float* __restrict__ out) {
  const int tr = blockIdx.x * 4 + (threadIdx.x >> 5);
  const int lane = threadIdx.x & 31;
  const int n = lane & 15, hf = lane >> 4;
  v8f acc[16];
#pragma unroll
  for (int f = 0; f < 16; ++f) acc[f] = zero8();
#if HAVE_ASYNC_LDS
  __shared__ __align__(16) _Float16 wbuf[2][16 * 512];  // 2 x 16KB
  stage_weights(w2h, wbuf[0]);
  for (int kk = 0; kk < 32; ++kk) {
    const int buf = kk & 1;
    v16h a = load_frag_swz(a1h + (size_t)(tr * 32 + kk) * 512);
    __builtin_amdgcn_s_wait_asynccnt(0);
    __syncthreads();
    if (kk + 1 < 32) stage_weights(w2h + (size_t)(kk + 1) * 16 * 512, wbuf[buf ^ 1]);
#pragma unroll
    for (int f = 0; f < 16; ++f)
      acc[f] = wmma16(a, load_frag_swz(&wbuf[buf][f * 512]), acc[f]);
    __syncthreads();
  }
#else
  for (int kk = 0; kk < 32; ++kk) {
    if (kk + 1 < 32) __builtin_prefetch(w2h + (size_t)(kk + 1) * 16 * 512, 0, 0);
    v16h a = load_frag_swz(a1h + (size_t)(tr * 32 + kk) * 512);
#pragma unroll
    for (int f = 0; f < 16; ++f)
      acc[f] = wmma16(a, load_frag_swz(w2h + (size_t)(kk * 16 + f) * 512), acc[f]);
  }
#endif
#pragma unroll
  for (int v = 0; v < 8; ++v) {
    float s = 0.f, q = 0.f;
#pragma unroll
    for (int f = 0; f < 16; ++f) {
      float t = acc[f][v] + b2[f * 16 + n];
      s += t; q += t * t;
    }
    s = rowsum16(s); q = rowsum16(q);
    const float mean = s * (1.0f / DD);
    const float var  = q * (1.0f / DD) - mean * mean;
    const float rstd = rsqrtf(var + LN_EPS);
    const int row = tr * 16 + v + 8 * hf;
#pragma unroll
    for (int f = 0; f < 16; ++f) {
      const int col = f * 16 + n;
      float t = acc[f][v] + b2[col];
      out[row * DD + col] = x[row * DD + col] + (t - mean) * rstd * g[col] + be[col];
    }
  }
}

// ---------------- launch ------------------------------------------------------

extern "C" void kernel_launch(void* const* d_in, const int* in_sizes, int n_in,
                              void* d_out, int out_size, void* d_ws, size_t ws_size,
                              hipStream_t stream) {
  const float* x      = (const float*)d_in[0];
  const float* wq     = (const float*)d_in[1];
  const float* wk     = (const float*)d_in[2];
  const float* wv     = (const float*)d_in[3];
  const float* w_proj = (const float*)d_in[4];
  const float* w1     = (const float*)d_in[5];
  const float* b1     = (const float*)d_in[6];
  const float* w2     = (const float*)d_in[7];
  const float* b2     = (const float*)d_in[8];
  const float* g1     = (const float*)d_in[9];
  const float* be1    = (const float*)d_in[10];
  const float* g2     = (const float*)d_in[11];
  const float* be2    = (const float*)d_in[12];
  float* out = (float*)d_out;

  char* ws = (char*)d_ws;
  auto take = [&](size_t elems) { _Float16* p = (_Float16*)ws; ws += elems * sizeof(_Float16); return p; };
  const size_t NX = (size_t)BB * SS * DD;       // 4,194,304
  const size_t NW = (size_t)HH * DD * EE;       // 65,536
  const size_t NQ = (size_t)BB * HH * SS * EE;  // 4,194,304
  _Float16* xh  = take(NX);
  _Float16* wqh = take(NW);
  _Float16* wkh = take(NW);
  _Float16* wvh = take(NW);
  _Float16* wph = take((size_t)DD * DD);
  _Float16* w1h = take((size_t)DD * FFD);
  _Float16* w2h = take((size_t)FFD * DD);
  _Float16* qh  = take(NQ);
  _Float16* kh  = take(NQ);
  _Float16* vh  = take(NQ);
  _Float16* ctxh = take(NX);
  _Float16* hh   = take(NX);
  _Float16* a1h  = take((size_t)BB * SS * FFD);

  cvt_a_swz_kernel<<<(int)((NX + 255) / 256), 256, 0, stream>>>(x, xh, DD, (int)NX);
  cvt_b_swz_kernel<<<(int)((NW + 255) / 256), 256, 0, stream>>>(wq, wqh, EE, (int)NW);
  cvt_b_swz_kernel<<<(int)((NW + 255) / 256), 256, 0, stream>>>(wk, wkh, EE, (int)NW);
  cvt_b_swz_kernel<<<(int)((NW + 255) / 256), 256, 0, stream>>>(wv, wvh, EE, (int)NW);
  cvt_b_swz_kernel<<<(DD * DD + 255) / 256, 256, 0, stream>>>(w_proj, wph, DD, DD * DD);
  cvt_b_swz_kernel<<<(DD * FFD + 255) / 256, 256, 0, stream>>>(w1, w1h, FFD, DD * FFD);
  cvt_b_swz_kernel<<<(FFD * DD + 255) / 256, 256, 0, stream>>>(w2, w2h, DD, FFD * DD);

  qkv_kernel<<<2048, 128, 0, stream>>>(xh, wqh, wkh, wvh, qh, kh, vh);
  attn_kernel<<<2048, 128, 0, stream>>>(qh, kh, vh, ctxh);
  proj_ln_kernel<<<256, 128, 0, stream>>>(ctxh, wph, x, g1, be1, hh);
  ff1_kernel<<<4096, 128, 0, stream>>>(hh, w1h, b1, a1h);
  ff2_ln_kernel<<<256, 128, 0, stream>>>(a1h, w2h, b2, x, g2, be2, out);

  (void)in_sizes; (void)n_in; (void)out_size; (void)ws_size;
}
